// Linear_mvm_52269751993097
// MI455X (gfx1250) — 8-bit matrix operand order repaired and checked by reading, compiled, not run on hardware
//
#include <hip/hip_runtime.h>
#include <stdint.h>


typedef __attribute__((ext_vector_type(8))) int v8i;

#define B_DIM 512
#define K_DIM 1024
#define N_DIM 1024
#define KT    (K_DIM / 64)
#define MT    (B_DIM / 16)
#define NT    (N_DIM / 16)

__device__ __forceinline__ int quant16(float v) {
    float r = rintf(v * 4096.0f);
    r = fminf(fmaxf(r, -32768.0f), 32767.0f);
    return (int)r;
}

__global__ void __launch_bounds__(256)
pack_a_kernel(const float* __restrict__ x,
              uint32_t* __restrict__ alo, uint32_t* __restrict__ ahi) {
    int t    = blockIdx.x * 256 + threadIdx.x;
    int lane = t & 31;
    int tile = t >> 5;
    int tM   = tile / KT;
    int tK   = tile - tM * KT;
    int m    = tM * 16 + (lane & 15);
    int half = lane >> 4;
    const float* row = x + (size_t)m * K_DIM + tK * 64;
    uint32_t* plo = alo + (size_t)t * 8;
    uint32_t* phi = ahi + (size_t)t * 8;
#pragma unroll
    for (int d = 0; d < 8; ++d) {
        int kb = ((d >> 1) * 16) + ((d & 1) * 4) + half * 8;
        uint32_t lo = 0u, hi = 0u;
#pragma unroll
        for (int j = 0; j < 4; ++j) {
            int q = quant16(row[kb + j]);
            lo |= (uint32_t)(q & 0xFF) << (8 * j);
            hi |= (uint32_t)((q >> 8) & 0xFF) << (8 * j);
        }
        plo[d] = lo;
        phi[d] = hi;
    }
}

__global__ void __launch_bounds__(256)
pack_b_kernel(const float* __restrict__ w,
              uint32_t* __restrict__ blo, uint32_t* __restrict__ bhi) {
    int t     = blockIdx.x * 256 + threadIdx.x;
    int lane  = t & 31;
    int tile  = t >> 5;
    int tN    = tile / KT;
    int tK    = tile - tN * KT;
    int n     = tN * 16 + (lane & 15);
    int khalf = lane >> 4;
    const float* row = w + (size_t)n * K_DIM + tK * 64;
    uint32_t* plo = blo + (size_t)t * 8;
    uint32_t* phi = bhi + (size_t)t * 8;
#pragma unroll
    for (int d = 0; d < 8; ++d) {
        int kb = ((d >> 1) * 16) + ((d & 1) * 4) + khalf * 8;
        uint32_t lo = 0u, hi = 0u;
#pragma unroll
        for (int j = 0; j < 4; ++j) {
            int q = quant16(row[kb + j]);
            lo |= (uint32_t)(q & 0xFF) << (8 * j);
            hi |= (uint32_t)((q >> 8) & 0xFF) << (8 * j);
        }
        plo[d] = lo;
        phi[d] = hi;
    }
}

__global__ void __launch_bounds__(256)
mvm_wmma_kernel(const uint32_t* __restrict__ alo, const uint32_t* __restrict__ ahi,
                const uint32_t* __restrict__ blo, const uint32_t* __restrict__ bhi,
                const float* __restrict__ bias, float* __restrict__ out) {
    const int wave = threadIdx.x >> 5;
    const int lane = threadIdx.x & 31;
    const int g    = blockIdx.x * 8 + wave;
    const int tM   = g / (NT / 2);
    const int gN   = g - tM * (NT / 2);
    const int tN0  = gN * 2;
    const int tN1  = tN0 + 1;

    const v8i* Alo = (const v8i*)alo;
    const v8i* Ahi = (const v8i*)ahi;
    const v8i* Blo = (const v8i*)blo;
    const v8i* Bhi = (const v8i*)bhi;

    v8i hh0 = {}; v8i hl0 = {}; v8i lh0 = {}; v8i ll0 = {};
    v8i hh1 = {}; v8i hl1 = {}; v8i lh1 = {}; v8i ll1 = {};

    int aidx  = (tM * KT) * 32 + lane;
    int bidx0 = (tN0 * KT) * 32 + lane;
    int bidx1 = (tN1 * KT) * 32 + lane;

#pragma unroll 4
    for (int tK = 0; tK < KT; ++tK) {
        v8i aL  = Alo[aidx + tK * 32];
        v8i aH  = Ahi[aidx + tK * 32];
        v8i bL0 = Blo[bidx0 + tK * 32];
        v8i bH0 = Bhi[bidx0 + tK * 32];
        v8i bL1 = Blo[bidx1 + tK * 32];
        v8i bH1 = Bhi[bidx1 + tK * 32];
        hh0 = __builtin_amdgcn_wmma_i32_16x16x64_iu8(true,  aH, true,  bH0, hh0, false, false);
        hl0 = __builtin_amdgcn_wmma_i32_16x16x64_iu8(true,  aH, false, bL0, hl0, false, false);
        lh0 = __builtin_amdgcn_wmma_i32_16x16x64_iu8(false, aL, true,  bH0, lh0, false, false);
        ll0 = __builtin_amdgcn_wmma_i32_16x16x64_iu8(false, aL, false, bL0, ll0, false, false);
        hh1 = __builtin_amdgcn_wmma_i32_16x16x64_iu8(true,  aH, true,  bH1, hh1, false, false);
        hl1 = __builtin_amdgcn_wmma_i32_16x16x64_iu8(true,  aH, false, bL1, hl1, false, false);
        lh1 = __builtin_amdgcn_wmma_i32_16x16x64_iu8(false, aL, true,  bH1, lh1, false, false);
        ll1 = __builtin_amdgcn_wmma_i32_16x16x64_iu8(false, aL, false, bL1, ll1, false, false);
    }

    const int n0  = tN0 * 16 + (lane & 15);
    const int n1  = tN1 * 16 + (lane & 15);
    const int m0  = tM * 16 + (lane >> 4) * 8;
    const float bv0 = bias[n0];
    const float bv1 = bias[n1];
#pragma unroll
    for (int v = 0; v < 8; ++v) {
        long long val0 = ((long long)hh0[v] << 16)
                       + (((long long)hl0[v] + (long long)lh0[v]) << 8)
                       + (long long)ll0[v];
        long long val1 = ((long long)hh1[v] << 16)
                       + (((long long)hl1[v] + (long long)lh1[v]) << 8)
                       + (long long)ll1[v];
        double r0 = rint((double)val0 * (1.0 / 4096.0));
        double r1 = rint((double)val1 * (1.0 / 4096.0));
        r0 = fmin(fmax(r0, -32768.0), 32767.0);
        r1 = fmin(fmax(r1, -32768.0), 32767.0);
        size_t rowoff = (size_t)(m0 + v) * N_DIM;
        out[rowoff + n0] = (float)r0 * (1.0f / 4096.0f) + bv0;
        out[rowoff + n1] = (float)r1 * (1.0f / 4096.0f) + bv1;
    }
}

extern "C" void kernel_launch(void* const* d_in, const int* in_sizes, int n_in,
                              void* d_out, int out_size, void* d_ws, size_t ws_size,
                              hipStream_t stream) {
    const float* x    = (const float*)d_in[0];
    const float* w    = (const float*)d_in[1];
    const float* bias = (const float*)d_in[2];
    float* out        = (float*)d_out;

    char* ws = (char*)d_ws;
    uint32_t* a_lo = (uint32_t*)(ws);
    uint32_t* a_hi = (uint32_t*)(ws + (size_t)512 * 1024);
    uint32_t* b_lo = (uint32_t*)(ws + (size_t)1024 * 1024);
    uint32_t* b_hi = (uint32_t*)(ws + (size_t)2048 * 1024);

    pack_a_kernel<<<(MT * KT * 32) / 256, 256, 0, stream>>>(x, a_lo, a_hi);
    pack_b_kernel<<<(NT * KT * 32) / 256, 256, 0, stream>>>(w, b_lo, b_hi);

    mvm_wmma_kernel<<<(MT * (NT / 2)) / 8, 256, 0, stream>>>(a_lo, a_hi, b_lo, b_hi, bias, out);
}
